// CWAB_34050500722860
// MI455X (gfx1250) — matrix operand order repaired and checked by reading, not compiled, not run on hardware
//
#include <hip/hip_runtime.h>
#include <hip/hip_bf16.h>
#include <stdint.h>

#define DEV static __device__ __forceinline__

// ---- problem constants -----------------------------------------------------
#define BB   4
#define SS   4096
#define DDim 1024
#define HH   16
#define HD   64
#define WINL 512
#define NWIN 8
#define GG   64
#define TWOG 128
#define SCL  0.125f          // 1/sqrt(64)

// ---- vector types ----------------------------------------------------------
typedef __attribute__((ext_vector_type(16))) __bf16        v16bf;
typedef __attribute__((ext_vector_type(16))) unsigned short v16u;
typedef __attribute__((ext_vector_type(8)))  unsigned short u16x8;
typedef __attribute__((ext_vector_type(8)))  float          v8f;
typedef __attribute__((ext_vector_type(4)))  float          f32x4;

// ---- helpers ---------------------------------------------------------------
DEV unsigned short f2bf(float f) {                 // round-to-nearest-even f32->bf16
  unsigned u = __float_as_uint(f);
  return (unsigned short)((u + 0x7FFFu + ((u >> 16) & 1u)) >> 16);
}

DEV v16bf as_bf16(v16u u) { union U { v16u u; v16bf b; } c; c.u = u; return c.b; }

// A-fragment: 16 contiguous elems (one row's K-chunk)
DEV v16u ld_a16(const unsigned short* p) {
  u16x8 lo = *(const u16x8*)p, hi = *(const u16x8*)(p + 8);
  v16u r;
#pragma unroll
  for (int i = 0; i < 8; ++i) { r[i] = lo[i]; r[i + 8] = hi[i]; }
  return r;
}
// B-fragment: 16 contiguous elems (one column's K-chunk)
DEV v16u ld_b16(const unsigned short* p) {
  u16x8 lo = *(const u16x8*)p, hi = *(const u16x8*)(p + 8);
  v16u r;
#pragma unroll
  for (int i = 0; i < 8; ++i) { r[i] = lo[i]; r[i + 8] = hi[i]; }
  return r;
}

DEV v8f wmma_bf(v16u a, v16u b, v8f c) {
  return __builtin_amdgcn_wmma_f32_16x16x32_bf16(false, as_bf16(a), false, as_bf16(b),
                                                 (short)0, c, false, false);
}

DEV void wait_ds0() { asm volatile("s_wait_dscnt 0" ::: "memory"); }

DEV float hmax16(float v) {
#pragma unroll
  for (int m = 1; m < 16; m <<= 1) v = fmaxf(v, __shfl_xor(v, m, 32));
  return v;
}
DEV float hsum16(float v) {
#pragma unroll
  for (int m = 1; m < 16; m <<= 1) v += __shfl_xor(v, m, 32);
  return v;
}

// ---- kernel 1: pos add + bf16 pack ----------------------------------------
__global__ void k_posadd(const float* __restrict__ x, const float* __restrict__ pos,
                         unsigned short* __restrict__ xb) {
  int i = blockIdx.x * 256 + threadIdx.x;                 // over B*S*D
  xb[i] = f2bf(x[i] + pos[i & (SS * DDim - 1)]);          // S*D power of two
}

// ---- prep kernels: bf16 "B-ready" weight transposes -----------------------
__global__ void k_prep_gatew(const float* __restrict__ gw, unsigned short* __restrict__ gwT) {
  int i = blockIdx.x * 256 + threadIdx.x;                 // over 2048*1024
  int c = i >> 10, o = i & 1023;                          // gw[c][o]
  gwT[o * 2048 + c] = f2bf(gw[i]);
}
__global__ void k_prep_outw(const float* __restrict__ ow, unsigned short* __restrict__ owT) {
  int i = blockIdx.x * 256 + threadIdx.x;                 // over 1024*1024
  int c = i >> 10, o = i & 1023;
  owT[o * 1024 + c] = f2bf(ow[i]);
}
__global__ void k_prep_convw(const float* __restrict__ cw, unsigned short* __restrict__ cwT) {
  int i = blockIdx.x * 256 + threadIdx.x;                 // over 1024*1024*4, cw[o][din][t]
  int t = i & 3, din = (i >> 2) & 1023, o = i >> 12;
  cwT[((size_t)(t * 1024 + o)) * 1024 + din] = f2bf(cw[i]);
}
__global__ void k_prep_gmem(const float* __restrict__ gm, unsigned short* __restrict__ gt) {
  int i = blockIdx.x * 256 + threadIdx.x;                 // over 64*1024
  unsigned short v = f2bf(gm[i]);
#pragma unroll
  for (int b = 0; b < BB; ++b) gt[(size_t)b * (TWOG * DDim) + GG * DDim + i] = v;
}

// ---- kernel: conv-as-GEMM -> gt rows 0..63 --------------------------------
// each wave computes a 16x64 strip (4 N tiles) to amortize the A fragment
__global__ __launch_bounds__(32) void k_conv(const unsigned short* __restrict__ xb,
                                             const unsigned short* __restrict__ cwT,
                                             const float* __restrict__ convb,
                                             unsigned short* __restrict__ gt) {
  const int nb = blockIdx.x, mt = blockIdx.y, b = blockIdx.z;   // nb: 0..15 (o strip of 64)
  const int lane = threadIdx.x, ln = lane & 15;
  const int ah = (lane < 16) ? 0 : 16, bh = (lane < 16) ? 0 : 16;
  const int hi8 = (lane < 16) ? 0 : 8;
  const int g = mt * 16 + ln;                 // A row (output token)
  const int ob = nb * 64;                     // strip base output channel
  v8f acc[4] = {{}, {}, {}, {}};
  for (int t = 0; t < 4; ++t) {
    const unsigned short* arow = xb + ((size_t)b * SS + 4 * g + t) * DDim;
    const unsigned short* bc0 = cwT + ((size_t)(t * 1024 + ob +  0 + ln)) * 1024;
    const unsigned short* bc1 = cwT + ((size_t)(t * 1024 + ob + 16 + ln)) * 1024;
    const unsigned short* bc2 = cwT + ((size_t)(t * 1024 + ob + 32 + ln)) * 1024;
    const unsigned short* bc3 = cwT + ((size_t)(t * 1024 + ob + 48 + ln)) * 1024;
    for (int dc = 0; dc < 1024; dc += 32) {
      v16u a = ld_a16(arow + dc + ah);
      acc[0] = wmma_bf(a, ld_b16(bc0 + dc + bh), acc[0]);
      acc[1] = wmma_bf(a, ld_b16(bc1 + dc + bh), acc[1]);
      acc[2] = wmma_bf(a, ld_b16(bc2 + dc + bh), acc[2]);
      acc[3] = wmma_bf(a, ld_b16(bc3 + dc + bh), acc[3]);
    }
  }
#pragma unroll
  for (int t = 0; t < 4; ++t) {
    const int o = ob + t * 16 + ln;
    const float cb = convb[o];
#pragma unroll
    for (int r = 0; r < 8; ++r) {
      int row = mt * 16 + r + hi8;
      gt[(size_t)b * (TWOG * DDim) + (size_t)row * DDim + o] = f2bf(acc[t][r] + cb);
    }
  }
}

// ---- flash attention kernels ----------------------------------------------
#define VTP 40          // LDS pitch for V^T (pad: stride 80B -> no bank clash, 16B aligned)
#define PTP 40          // LDS pitch for per-wave P tile

__global__ __launch_bounds__(256) void k_local(const unsigned short* __restrict__ xb,
                                               float* __restrict__ localb,
                                               unsigned short* __restrict__ localbf) {
  __shared__ __align__(16) unsigned short Vt[64 * VTP];
  __shared__ __align__(16) unsigned short Pt[8 * 16 * PTP];
  const int qb = blockIdx.x, wh = blockIdx.y, b = blockIdx.z;
  const int w = wh >> 4, h = wh & 15;
  const int tid = threadIdx.x, wave = tid >> 5, lane = tid & 31, ln = lane & 15;
  const int ah = (lane < 16) ? 0 : 16, bh = (lane < 16) ? 0 : 16;
  const int hi8 = (lane < 16) ? 0 : 8;
  const unsigned short* xbb = xb + (size_t)b * SS * DDim;
  const int qrow0 = w * WINL + qb * 128 + wave * 16;
  const unsigned short* qp = xbb + (size_t)(qrow0 + ln) * DDim + h * HD;
  v16u qa0 = ld_a16(qp + ah);
  v16u qa1 = ld_a16(qp + 32 + ah);
  v8f o0 = {}, o1 = {}, o2 = {}, o3 = {};
  float m[8], l[8];
#pragma unroll
  for (int r = 0; r < 8; ++r) { m[r] = -1e30f; l[r] = 0.f; }
  unsigned short* myPt = Pt + wave * (16 * PTP);

  for (int kc = 0; kc < 16; ++kc) {
    const int kb = w * WINL + kc * 32;
    {  // cooperative V^T staging: Vt[d][k] = x[kb+k][h*64+d]
      int d = tid & 63, k0 = (tid >> 6) * 8;
#pragma unroll
      for (int i = 0; i < 8; ++i)
        Vt[d * VTP + k0 + i] = xbb[(size_t)(kb + k0 + i) * DDim + h * HD + d];
    }
    __syncthreads();
    const unsigned short* kp0 = xbb + (size_t)(kb + ln) * DDim + h * HD;
    const unsigned short* kp1 = xbb + (size_t)(kb + 16 + ln) * DDim + h * HD;
    v8f s0 = {}, s1 = {};
    s0 = wmma_bf(qa0, ld_b16(kp0 + bh), s0);
    s0 = wmma_bf(qa1, ld_b16(kp0 + 32 + bh), s0);
    s1 = wmma_bf(qa0, ld_b16(kp1 + bh), s1);
    s1 = wmma_bf(qa1, ld_b16(kp1 + 32 + bh), s1);
#pragma unroll
    for (int r = 0; r < 8; ++r) {
      float sa = s0[r] * SCL, sb = s1[r] * SCL;
      float mn = fmaxf(m[r], hmax16(fmaxf(sa, sb)));
      float f  = __expf(m[r] - mn);
      float pa = __expf(sa - mn), pb = __expf(sb - mn);
      l[r] = l[r] * f + hsum16(pa + pb);
      m[r] = mn;
      o0[r] *= f; o1[r] *= f; o2[r] *= f; o3[r] *= f;
      const int row = r + hi8;
      myPt[row * PTP + ln]      = f2bf(pa);
      myPt[row * PTP + 16 + ln] = f2bf(pb);
    }
    wait_ds0();
    v16u pfr = ld_a16(myPt + ln * PTP + ah);
    o0 = wmma_bf(pfr, ld_b16(Vt + (0 * 16 + ln) * VTP + bh), o0);
    o1 = wmma_bf(pfr, ld_b16(Vt + (1 * 16 + ln) * VTP + bh), o1);
    o2 = wmma_bf(pfr, ld_b16(Vt + (2 * 16 + ln) * VTP + bh), o2);
    o3 = wmma_bf(pfr, ld_b16(Vt + (3 * 16 + ln) * VTP + bh), o3);
    __syncthreads();
  }
#pragma unroll
  for (int r = 0; r < 8; ++r) {
    const int row = qrow0 + r + hi8;
    const float inv = 1.f / l[r];
    const size_t base = ((size_t)b * SS + row) * DDim + h * HD + ln;
    float v0 = o0[r] * inv, v1 = o1[r] * inv, v2 = o2[r] * inv, v3 = o3[r] * inv;
    localb[base + 0] = v0; localb[base + 16] = v1;
    localb[base + 32] = v2; localb[base + 48] = v3;
    localbf[base + 0] = f2bf(v0); localbf[base + 16] = f2bf(v1);
    localbf[base + 32] = f2bf(v2); localbf[base + 48] = f2bf(v3);
  }
}

__global__ __launch_bounds__(256) void k_glob(const unsigned short* __restrict__ xb,
                                              const unsigned short* __restrict__ gt,
                                              float* __restrict__ globb,
                                              unsigned short* __restrict__ globbf) {
  __shared__ __align__(16) unsigned short Vt[64 * VTP];
  __shared__ __align__(16) unsigned short Pt[8 * 16 * PTP];
  const int sb = blockIdx.x, h = blockIdx.y, b = blockIdx.z;
  const int tid = threadIdx.x, wave = tid >> 5, lane = tid & 31, ln = lane & 15;
  const int ah = (lane < 16) ? 0 : 16, bh = (lane < 16) ? 0 : 16;
  const int hi8 = (lane < 16) ? 0 : 8;
  const unsigned short* xbb = xb + (size_t)b * SS * DDim;
  const unsigned short* gtb = gt + (size_t)b * TWOG * DDim;
  const int qrow0 = sb * 128 + wave * 16;
  const unsigned short* qp = xbb + (size_t)(qrow0 + ln) * DDim + h * HD;
  v16u qa0 = ld_a16(qp + ah);
  v16u qa1 = ld_a16(qp + 32 + ah);
  v8f o0 = {}, o1 = {}, o2 = {}, o3 = {};
  float m[8], l[8];
#pragma unroll
  for (int r = 0; r < 8; ++r) { m[r] = -1e30f; l[r] = 0.f; }
  unsigned short* myPt = Pt + wave * (16 * PTP);

  for (int kc = 0; kc < 4; ++kc) {            // 128 global tokens, chunks of 32
    const int kb = kc * 32;
    {
      int d = tid & 63, k0 = (tid >> 6) * 8;
#pragma unroll
      for (int i = 0; i < 8; ++i)
        Vt[d * VTP + k0 + i] = gtb[(size_t)(kb + k0 + i) * DDim + h * HD + d];
    }
    __syncthreads();
    const unsigned short* kp0 = gtb + (size_t)(kb + ln) * DDim + h * HD;
    const unsigned short* kp1 = gtb + (size_t)(kb + 16 + ln) * DDim + h * HD;
    v8f s0 = {}, s1 = {};
    s0 = wmma_bf(qa0, ld_b16(kp0 + bh), s0);
    s0 = wmma_bf(qa1, ld_b16(kp0 + 32 + bh), s0);
    s1 = wmma_bf(qa0, ld_b16(kp1 + bh), s1);
    s1 = wmma_bf(qa1, ld_b16(kp1 + 32 + bh), s1);
#pragma unroll
    for (int r = 0; r < 8; ++r) {
      float sa = s0[r] * SCL, sb = s1[r] * SCL;
      float mn = fmaxf(m[r], hmax16(fmaxf(sa, sb)));
      float f  = __expf(m[r] - mn);
      float pa = __expf(sa - mn), pb = __expf(sb - mn);
      l[r] = l[r] * f + hsum16(pa + pb);
      m[r] = mn;
      o0[r] *= f; o1[r] *= f; o2[r] *= f; o3[r] *= f;
      const int row = r + hi8;
      myPt[row * PTP + ln]      = f2bf(pa);
      myPt[row * PTP + 16 + ln] = f2bf(pb);
    }
    wait_ds0();
    v16u pfr = ld_a16(myPt + ln * PTP + ah);
    o0 = wmma_bf(pfr, ld_b16(Vt + (0 * 16 + ln) * VTP + bh), o0);
    o1 = wmma_bf(pfr, ld_b16(Vt + (1 * 16 + ln) * VTP + bh), o1);
    o2 = wmma_bf(pfr, ld_b16(Vt + (2 * 16 + ln) * VTP + bh), o2);
    o3 = wmma_bf(pfr, ld_b16(Vt + (3 * 16 + ln) * VTP + bh), o3);
    __syncthreads();
  }
#pragma unroll
  for (int r = 0; r < 8; ++r) {
    const int row = qrow0 + r + hi8;
    const float inv = 1.f / l[r];
    const size_t base = ((size_t)b * SS + row) * DDim + h * HD + ln;
    float v0 = o0[r] * inv, v1 = o1[r] * inv, v2 = o2[r] * inv, v3 = o3[r] * inv;
    globb[base + 0] = v0; globb[base + 16] = v1;
    globb[base + 32] = v2; globb[base + 48] = v3;
    globbf[base + 0] = f2bf(v0); globbf[base + 16] = f2bf(v1);
    globbf[base + 32] = f2bf(v2); globbf[base + 48] = f2bf(v3);
  }
}

// ---- gate GEMM + sigmoid mix (16x64 strip per wave, bf16 A) ----------------
__global__ __launch_bounds__(32) void k_gate(const float* __restrict__ localb,
                                             const float* __restrict__ globb,
                                             const unsigned short* __restrict__ localbf,
                                             const unsigned short* __restrict__ globbf,
                                             const unsigned short* __restrict__ gwT,
                                             const float* __restrict__ gateb,
                                             unsigned short* __restrict__ ybf) {
  const int nb = blockIdx.x, mt = blockIdx.y;    // nb: 0..15 (o strip of 64)
  const int lane = threadIdx.x, ln = lane & 15;
  const int ah = (lane < 16) ? 0 : 16, bh = (lane < 16) ? 0 : 16;
  const int hi8 = (lane < 16) ? 0 : 8;
  const int row = mt * 16 + ln;                  // flat token row (B*S)
  const int ob  = nb * 64;                       // strip base output channel
  v8f acc[4] = {{}, {}, {}, {}};
  const unsigned short* bc0 = gwT + (size_t)(ob +  0 + ln) * 2048;
  const unsigned short* bc1 = gwT + (size_t)(ob + 16 + ln) * 2048;
  const unsigned short* bc2 = gwT + (size_t)(ob + 32 + ln) * 2048;
  const unsigned short* bc3 = gwT + (size_t)(ob + 48 + ln) * 2048;
  const unsigned short* arow = localbf + (size_t)row * DDim;
  for (int c = 0; c < 1024; c += 32) {
    v16u a = ld_a16(arow + c + ah);
    acc[0] = wmma_bf(a, ld_b16(bc0 + c + bh), acc[0]);
    acc[1] = wmma_bf(a, ld_b16(bc1 + c + bh), acc[1]);
    acc[2] = wmma_bf(a, ld_b16(bc2 + c + bh), acc[2]);
    acc[3] = wmma_bf(a, ld_b16(bc3 + c + bh), acc[3]);
  }
  arow = globbf + (size_t)row * DDim;
  for (int c = 0; c < 1024; c += 32) {
    v16u a = ld_a16(arow + c + ah);
    acc[0] = wmma_bf(a, ld_b16(bc0 + 1024 + c + bh), acc[0]);
    acc[1] = wmma_bf(a, ld_b16(bc1 + 1024 + c + bh), acc[1]);
    acc[2] = wmma_bf(a, ld_b16(bc2 + 1024 + c + bh), acc[2]);
    acc[3] = wmma_bf(a, ld_b16(bc3 + 1024 + c + bh), acc[3]);
  }
#pragma unroll
  for (int t = 0; t < 4; ++t) {
    const int o = ob + t * 16 + ln;
    const float gb = gateb[o];
#pragma unroll
    for (int r = 0; r < 8; ++r) {
      const int rr = mt * 16 + r + hi8;
      float mix = 1.f / (1.f + __expf(-(acc[t][r] + gb)));
      float lo = localb[(size_t)rr * DDim + o];
      float gl = globb[(size_t)rr * DDim + o];
      ybf[(size_t)rr * DDim + o] = f2bf(mix * lo + (1.f - mix) * gl);
    }
  }
}

// ---- output projection (16x64 strip per wave) ------------------------------
__global__ __launch_bounds__(32) void k_out(const unsigned short* __restrict__ ybf,
                                            const unsigned short* __restrict__ owT,
                                            const float* __restrict__ outb,
                                            float* __restrict__ out) {
  const int nb = blockIdx.x, mt = blockIdx.y;    // nb: 0..15
  const int lane = threadIdx.x, ln = lane & 15;
  const int ah = (lane < 16) ? 0 : 16, bh = (lane < 16) ? 0 : 16;
  const int hi8 = (lane < 16) ? 0 : 8;
  const int row = mt * 16 + ln;
  const int ob  = nb * 64;
  v8f acc[4] = {{}, {}, {}, {}};
  const unsigned short* arow = ybf + (size_t)row * DDim;
  const unsigned short* bc0 = owT + (size_t)(ob +  0 + ln) * 1024;
  const unsigned short* bc1 = owT + (size_t)(ob + 16 + ln) * 1024;
  const unsigned short* bc2 = owT + (size_t)(ob + 32 + ln) * 1024;
  const unsigned short* bc3 = owT + (size_t)(ob + 48 + ln) * 1024;
  for (int c = 0; c < 1024; c += 32) {
    v16u a = ld_a16(arow + c + ah);
    acc[0] = wmma_bf(a, ld_b16(bc0 + c + bh), acc[0]);
    acc[1] = wmma_bf(a, ld_b16(bc1 + c + bh), acc[1]);
    acc[2] = wmma_bf(a, ld_b16(bc2 + c + bh), acc[2]);
    acc[3] = wmma_bf(a, ld_b16(bc3 + c + bh), acc[3]);
  }
#pragma unroll
  for (int t = 0; t < 4; ++t) {
    const int o = ob + t * 16 + ln;
    const float obi = outb[o];
#pragma unroll
    for (int r = 0; r < 8; ++r) {
      const int rr = mt * 16 + r + hi8;
      out[(size_t)rr * DDim + o] = acc[t][r] + obi;
    }
  }
}

// ---- host launch -----------------------------------------------------------
extern "C" void kernel_launch(void* const* d_in, const int* in_sizes, int n_in,
                              void* d_out, int out_size, void* d_ws, size_t ws_size,
                              hipStream_t stream) {
  (void)in_sizes; (void)n_in; (void)out_size; (void)ws_size;
  const float* x     = (const float*)d_in[0];
  const float* pos   = (const float*)d_in[1];
  const float* gmem  = (const float*)d_in[2];
  const float* convw = (const float*)d_in[3];
  const float* convb = (const float*)d_in[4];
  const float* gatew = (const float*)d_in[5];
  const float* gateb = (const float*)d_in[6];
  const float* outw  = (const float*)d_in[7];
  const float* outb  = (const float*)d_in[8];
  float* out = (float*)d_out;

  char* ws = (char*)d_ws;
  size_t off = 0;
  auto alloc = [&](size_t bytes) {
    char* p = ws + off;
    off += (bytes + 255) & ~(size_t)255;
    return p;
  };
  unsigned short* xb   = (unsigned short*)alloc((size_t)BB * SS * DDim * 2);   // 32 MB
  float* localb        = (float*)alloc((size_t)BB * SS * DDim * 4);            // 64 MB
  float* globb         = (float*)alloc((size_t)BB * SS * DDim * 4);            // 64 MB
  unsigned short* lbf  = (unsigned short*)alloc((size_t)BB * SS * DDim * 2);   // 32 MB
  unsigned short* gbf  = (unsigned short*)alloc((size_t)BB * SS * DDim * 2);   // 32 MB
  unsigned short* ybf  = (unsigned short*)alloc((size_t)BB * SS * DDim * 2);   // 32 MB
  unsigned short* gtb  = (unsigned short*)alloc((size_t)BB * TWOG * DDim * 2); // 1 MB
  unsigned short* gwT  = (unsigned short*)alloc((size_t)2 * DDim * DDim * 2);  // 4 MB
  unsigned short* owT  = (unsigned short*)alloc((size_t)DDim * DDim * 2);      // 2 MB
  unsigned short* cwT  = (unsigned short*)alloc((size_t)4 * DDim * DDim * 2);  // 8 MB

  k_posadd    <<<65536, 256, 0, stream>>>(x, pos, xb);
  k_prep_gatew<<<8192,  256, 0, stream>>>(gatew, gwT);
  k_prep_outw <<<4096,  256, 0, stream>>>(outw, owT);
  k_prep_convw<<<16384, 256, 0, stream>>>(convw, cwT);
  k_prep_gmem <<<256,   256, 0, stream>>>(gmem, gtb);
  k_conv <<<dim3(16, 4, 4),          32, 0, stream>>>(xb, cwT, convb, gtb);
  k_local<<<dim3(4, NWIN * HH, BB), 256, 0, stream>>>(xb, localb, lbf);
  k_glob <<<dim3(SS / 128, HH, BB), 256, 0, stream>>>(xb, gtb, globb, gbf);
  k_gate <<<dim3(16, (BB * SS) / 16), 32, 0, stream>>>(localb, globb, lbf, gbf, gwT, gateb, ybf);
  k_out  <<<dim3(16, (BB * SS) / 16), 32, 0, stream>>>(ybf, owT, outb, out);
}
